// MultiheadSelfAttention_29712583754262
// MI455X (gfx1250) — matrix operand order repaired and checked by reading, not compiled, not run on hardware
//
#include <hip/hip_runtime.h>

#define DIM 768
#define SEQ 4096
#define NH  12
#define HD  64
#define QK_SCALE 0.125f   // 64^-0.5

typedef __attribute__((ext_vector_type(16))) __bf16       v16bf;
typedef __attribute__((ext_vector_type(8)))  float        v8f;
typedef __attribute__((ext_vector_type(8)))  unsigned int v8u;
typedef __attribute__((ext_vector_type(4)))  unsigned int v4u;
typedef __attribute__((ext_vector_type(8)))  int          v8i;
typedef __attribute__((ext_vector_type(4)))  int          v4i;

// ---------- helpers ----------

static __device__ __forceinline__ unsigned int f2bf(float f) {
  unsigned int u = __float_as_uint(f);
  return (u + 0x7FFFu + ((u >> 16) & 1u)) >> 16;   // RNE f32 -> bf16 bits
}
static __device__ __forceinline__ unsigned int pack2bf(float lo, float hi) {
  return f2bf(lo) | (f2bf(hi) << 16);
}

// A-fragment, 16x32 bf16, from row-major matrix (row stride ld elements).
// lane m=lane&15 holds row m; hi=lane>>4; dwords 0..3 = K 8hi+{0..7},
// dwords 4..7 = K 16+8hi+{0..7} -> two 16B loads (global or ds).
static __device__ __forceinline__ v16bf
load_a_frag(const unsigned short* base, int row0, int k0, int ld, int lane) {
  int m = lane & 15, hi = lane >> 4;
  const unsigned short* r = base + (size_t)(row0 + m) * ld + k0 + 8 * hi;
  v4u lo = *(const v4u*)r;
  v4u hh = *(const v4u*)(r + 16);
  v8u f;
  f[0] = lo[0]; f[1] = lo[1]; f[2] = lo[2]; f[3] = lo[3];
  f[4] = hh[0]; f[5] = hh[1]; f[6] = hh[2]; f[7] = hh[3];
  return __builtin_bit_cast(v16bf, f);
}

// B-fragment, 32x16 bf16: lane n=lane&15 holds column n; hi=lane>>4;
// dwords 0..3 = K 8hi+{0..7}, dwords 4..7 = K 16+8hi+{0..7}.
static __device__ __forceinline__ v16bf
load_b_frag(const unsigned short* base, int k0, int n0, int ld, int lane) {
  int n = lane & 15, hi = lane >> 4;
  const unsigned short* r = base + (size_t)(k0 + 8 * hi) * ld + n0 + n;
  v8u f;
#pragma unroll
  for (int j = 0; j < 4; ++j) {
    f[j]     = (unsigned int)r[(size_t)(2 * j) * ld] |
               ((unsigned int)r[(size_t)(2 * j + 1) * ld] << 16);
    f[4 + j] = (unsigned int)r[(size_t)(16 + 2 * j) * ld] |
               ((unsigned int)r[(size_t)(17 + 2 * j) * ld] << 16);
  }
  return __builtin_bit_cast(v16bf, f);
}

static __device__ __forceinline__ v8f wmma_bf16(v16bf a, v16bf b, v8f c) {
  return __builtin_amdgcn_wmma_f32_16x16x32_bf16(
      false, a, false, b, (short)0, c, false, false);
}

// ---------- conversion kernels ----------

__global__ void k_cvt1d(const float* __restrict__ src,
                        unsigned short* __restrict__ dst, int n) {
  int i = blockIdx.x * blockDim.x + threadIdx.x;
  int stride = gridDim.x * blockDim.x;
  for (; i < n; i += stride) dst[i] = (unsigned short)f2bf(src[i]);
}

__global__ void k_cvtT(const float* __restrict__ src,
                       unsigned short* __restrict__ dst, int rows, int cols) {
  int i = blockIdx.x * blockDim.x + threadIdx.x;
  if (i >= rows * cols) return;
  int r = i / cols, c = i - r * cols;
  dst[(size_t)c * rows + r] = (unsigned short)f2bf(src[i]);
}

// ---------- QKV projection GEMM ----------
// One wave computes a 32(s) x 64(e) tile: two A-frags share each B-frag.
__global__ __launch_bounds__(128) void
k_qkv(const unsigned short* __restrict__ xb, const unsigned short* __restrict__ wT,
      unsigned short* __restrict__ Qt, unsigned short* __restrict__ Kb,
      unsigned short* __restrict__ Vb) {
  int lane = threadIdx.x & 31;
  int wid  = blockIdx.x * 4 + (threadIdx.x >> 5);
  int st = wid / 36, et = wid % 36;     // st in [0,128)
  int s0 = st * 32, e0 = et * 64;

  v8f acc[2][4];
#pragma unroll
  for (int r = 0; r < 2; ++r)
#pragma unroll
    for (int t = 0; t < 4; ++t)
#pragma unroll
      for (int j = 0; j < 8; ++j) acc[r][t][j] = 0.f;

  for (int k0 = 0; k0 < DIM; k0 += 32) {
    v16bf a0 = load_a_frag(xb, s0,      k0, DIM, lane);
    v16bf a1 = load_a_frag(xb, s0 + 16, k0, DIM, lane);
#pragma unroll
    for (int t = 0; t < 4; ++t) {
      v16bf b = load_b_frag(wT, k0, e0 + 16 * t, 3 * DIM, lane);
      acc[0][t] = wmma_bf16(a0, b, acc[0][t]);
      acc[1][t] = wmma_bf16(a1, b, acc[1][t]);
    }
  }

  int sel = et / 12, h = et % 12;
  int n = lane & 15, hi = lane >> 4;
#pragma unroll
  for (int r = 0; r < 2; ++r) {
    int sr = s0 + 16 * r;
    if (sel == 0) {
      // Q: pre-scale, store transposed d-major: Qt[h][d][s]
#pragma unroll
      for (int t = 0; t < 4; ++t) {
        unsigned short* qp = Qt + ((size_t)h * HD + t * 16 + n) * SEQ + sr + 8 * hi;
#pragma unroll
        for (int j = 0; j < 8; j += 2)
          *(unsigned int*)(qp + j) =
              pack2bf(acc[r][t][j] * QK_SCALE, acc[r][t][j + 1] * QK_SCALE);
      }
    } else {
      unsigned short* dst = (sel == 1 ? Kb : Vb) + (size_t)h * SEQ * HD;
#pragma unroll
      for (int t = 0; t < 4; ++t)
#pragma unroll
        for (int j = 0; j < 8; ++j)
          dst[(size_t)(sr + j + 8 * hi) * HD + t * 16 + n] =
              (unsigned short)f2bf(acc[r][t][j]);
    }
  }
}

// ---------- flash attention with TDM double-buffered K/V in LDS ----------
// Block = 4 waves = 64 consecutive queries of one head. Wave 0 drives the
// Tensor Data Mover: one tensor_load_to_lds per 32x64 bf16 K (and V) tile,
// double buffered, gated by s_wait_tensorcnt; fragments come from LDS.
__global__ __launch_bounds__(128) void
k_attn(const unsigned short* __restrict__ Qt, const unsigned short* __restrict__ Kb,
       const unsigned short* __restrict__ Vb, unsigned short* __restrict__ attnb) {
  __shared__ __attribute__((aligned(16))) unsigned short kbuf[2][32 * HD];
  __shared__ __attribute__((aligned(16))) unsigned short vbuf[2][32 * HD];

  int lane = threadIdx.x & 31;
  int wv   = threadIdx.x >> 5;
  int h    = blockIdx.x / 64;
  int q0   = (blockIdx.x % 64) * 64 + wv * 16;
  int hi = lane >> 4, n = lane & 15;

  const unsigned short* Qh = Qt + (size_t)h * HD * SEQ;  // [d][s]
  const unsigned short* Kh = Kb + (size_t)h * SEQ * HD;  // [s][d]
  const unsigned short* Vh = Vb + (size_t)h * SEQ * HD;  // [s][d]

  // D# group1: workgroup_mask=0, data_size=2B, tensor_dim0=64,
  // tensor_dim1=4096, tile_dim0=64, tile_dim1=32, tensor_dim0_stride=64.
  v8i g1 = {0x00010000, 0x00400000, 0x10000000, 0x00400000, 32, 64, 0, 0};
  v4i gz4 = {0, 0, 0, 0};
  v8i gz8 = {0, 0, 0, 0, 0, 0, 0, 0};
  auto tdm_load = [&](const unsigned short* gptr, unsigned short* lptr) {
    unsigned long long ga = (unsigned long long)(uintptr_t)gptr;
    unsigned lds = (unsigned)(uintptr_t)lptr;  // low 32 bits = LDS byte offset
    v4u g0;
    g0[0] = 1u;                                // count=1, user descriptor
    g0[1] = lds;                               // lds_addr [63:32]
    g0[2] = (unsigned)ga;                      // global_addr [95:64]
    g0[3] = ((unsigned)(ga >> 32) & 0x01FFFFFFu) | (2u << 30);  // addr hi | type=2
    __builtin_amdgcn_tensor_load_to_lds(g0, g1, gz4, gz4, gz8, 0);
  };

  // Q^T B-fragments, loaded once from global (lane = query, elements = d)
  v16bf qf0 = load_b_frag(Qh, 0, q0, SEQ, lane);
  v16bf qf1 = load_b_frag(Qh, 32, q0, SEQ, lane);

  v8f o[4];
#pragma unroll
  for (int t = 0; t < 4; ++t)
#pragma unroll
    for (int j = 0; j < 8; ++j) o[t][j] = 0.f;
  float mrun = -1e30f, lrun = 0.f;

  if (wv == 0) { tdm_load(Kh, &kbuf[0][0]); tdm_load(Vh, &vbuf[0][0]); }

  for (int it = 0; it < SEQ / 32; ++it) {
    int key0 = it * 32;
    if (wv == 0) {
      if (it + 1 < SEQ / 32) {
        tdm_load(Kh + (size_t)(key0 + 32) * HD, &kbuf[(it + 1) & 1][0]);
        tdm_load(Vh + (size_t)(key0 + 32) * HD, &vbuf[(it + 1) & 1][0]);
        __builtin_amdgcn_s_wait_tensorcnt(2);  // current buffer complete
      } else {
        __builtin_amdgcn_s_wait_tensorcnt(0);
      }
    }
    __syncthreads();  // publish current K/V tile
    const unsigned short* kp = &kbuf[it & 1][0];
    const unsigned short* vp = &vbuf[it & 1][0];

    v16bf ka00 = load_a_frag(kp, 0,  0,  HD, lane);
    v16bf ka01 = load_a_frag(kp, 0,  32, HD, lane);
    v16bf ka10 = load_a_frag(kp, 16, 0,  HD, lane);
    v16bf ka11 = load_a_frag(kp, 16, 32, HD, lane);

    v8f t0, t1;
#pragma unroll
    for (int j = 0; j < 8; ++j) { t0[j] = 0.f; t1[j] = 0.f; }
    t0 = wmma_bf16(ka00, qf0, t0); t0 = wmma_bf16(ka01, qf1, t0);
    t1 = wmma_bf16(ka10, qf0, t1); t1 = wmma_bf16(ka11, qf1, t1);
    // S^T layout: lane&15 = query, vgpr j = key (j + 8*hi) per 16-tile

    float mx = -1e30f;
#pragma unroll
    for (int j = 0; j < 8; ++j) mx = fmaxf(mx, fmaxf(t0[j], t1[j]));
    mx = fmaxf(mx, __shfl_xor(mx, 16, 32));
    float mnew  = fmaxf(mrun, mx);
    float alpha = __expf(mrun - mnew);
    float rs = 0.f;
#pragma unroll
    for (int j = 0; j < 8; ++j) {
      float p0 = __expf(t0[j] - mnew); t0[j] = p0;
      float p1 = __expf(t1[j] - mnew); t1[j] = p1;
      rs += p0 + p1;
    }
    rs += __shfl_xor(rs, 16, 32);
    lrun = lrun * alpha + rs;
    mrun = mnew;

    // In-lane repack: P (16q x 32k) A-fragment from S^T C/D values.
    v8u pu;
#pragma unroll
    for (int j = 0; j < 4; ++j) {
      pu[j]     = pack2bf(t0[2 * j], t0[2 * j + 1]);
      pu[4 + j] = pack2bf(t1[2 * j], t1[2 * j + 1]);
    }
    v16bf pf = __builtin_bit_cast(v16bf, pu);

    float arow[8];
#pragma unroll
    for (int j = 0; j < 8; ++j) arow[j] = __shfl(alpha, j + 8 * hi, 32);

#pragma unroll
    for (int t = 0; t < 4; ++t) {
      v16bf vf = load_b_frag(vp, 0, t * 16, HD, lane);  // lane = d, elements = keys
#pragma unroll
      for (int j = 0; j < 8; ++j) o[t][j] *= arow[j];
      o[t] = wmma_bf16(pf, vf, o[t]);
    }
    __syncthreads();  // all waves done with this buffer before it is reloaded
  }

  float linv[8];
#pragma unroll
  for (int j = 0; j < 8; ++j) linv[j] = 1.f / __shfl(lrun, j + 8 * hi, 32);
#pragma unroll
  for (int t = 0; t < 4; ++t)
#pragma unroll
    for (int j = 0; j < 8; ++j)
      attnb[(size_t)(q0 + j + 8 * hi) * DIM + h * HD + t * 16 + n] =
          (unsigned short)f2bf(o[t][j] * linv[j]);
}

// ---------- output projection GEMM (32x64 tile per wave) ----------
__global__ __launch_bounds__(128) void
k_proj(const unsigned short* __restrict__ attnb, const unsigned short* __restrict__ wT,
       const float* __restrict__ bias, float* __restrict__ out) {
  int lane = threadIdx.x & 31;
  int wid  = blockIdx.x * 4 + (threadIdx.x >> 5);
  int st = wid / 12, et = wid % 12;     // st in [0,128)
  int s0 = st * 32, e0 = et * 64;

  v8f acc[2][4];
#pragma unroll
  for (int r = 0; r < 2; ++r)
#pragma unroll
    for (int t = 0; t < 4; ++t)
#pragma unroll
      for (int j = 0; j < 8; ++j) acc[r][t][j] = 0.f;

  for (int k0 = 0; k0 < DIM; k0 += 32) {
    v16bf a0 = load_a_frag(attnb, s0,      k0, DIM, lane);
    v16bf a1 = load_a_frag(attnb, s0 + 16, k0, DIM, lane);
#pragma unroll
    for (int t = 0; t < 4; ++t) {
      v16bf b = load_b_frag(wT, k0, e0 + 16 * t, DIM, lane);
      acc[0][t] = wmma_bf16(a0, b, acc[0][t]);
      acc[1][t] = wmma_bf16(a1, b, acc[1][t]);
    }
  }

  int nn = lane & 15, hi = lane >> 4;
#pragma unroll
  for (int r = 0; r < 2; ++r) {
    int sr = s0 + 16 * r;
#pragma unroll
    for (int t = 0; t < 4; ++t) {
      float bv = bias[e0 + t * 16 + nn];
#pragma unroll
      for (int j = 0; j < 8; ++j)
        out[(size_t)(sr + j + 8 * hi) * DIM + e0 + t * 16 + nn] = acc[r][t][j] + bv;
    }
  }
}

// ---------- launcher ----------

extern "C" void kernel_launch(void* const* d_in, const int* in_sizes, int n_in,
                              void* d_out, int out_size, void* d_ws, size_t ws_size,
                              hipStream_t stream) {
  (void)in_sizes; (void)n_in; (void)out_size; (void)ws_size;
  const float* x      = (const float*)d_in[0];
  const float* w_qkv  = (const float*)d_in[1];
  const float* w_proj = (const float*)d_in[2];
  const float* b_proj = (const float*)d_in[3];
  float* out = (float*)d_out;

  unsigned short* xb     = (unsigned short*)d_ws;
  unsigned short* wqkvT  = xb     + (size_t)SEQ * DIM;        // [768][2304]
  unsigned short* wprojT = wqkvT  + (size_t)DIM * 3 * DIM;    // [768][768]
  unsigned short* Qt     = wprojT + (size_t)DIM * DIM;        // [12][64][4096]
  unsigned short* Kb     = Qt     + (size_t)NH * SEQ * HD;    // [12][4096][64]
  unsigned short* Vb     = Kb     + (size_t)NH * SEQ * HD;    // [12][4096][64]
  unsigned short* attnb  = Vb     + (size_t)NH * SEQ * HD;    // [4096][768]

  k_cvt1d<<<2048, 256, 0, stream>>>(x, xb, SEQ * DIM);
  k_cvtT<<<(3 * DIM * DIM + 255) / 256, 256, 0, stream>>>(w_qkv, wqkvT, 3 * DIM, DIM);
  k_cvtT<<<(DIM * DIM + 255) / 256, 256, 0, stream>>>(w_proj, wprojT, DIM, DIM);
  k_qkv <<<(128 * 36) / 4, 128, 0, stream>>>(xb, wqkvT, Qt, Kb, Vb);
  k_attn<<<(NH * 64), 128, 0, stream>>>(Qt, Kb, Vb, attnb);
  k_proj<<<(128 * 12) / 4, 128, 0, stream>>>(attnb, wprojT, b_proj, out);
}
